// WorkingMemory_20976620274274
// MI455X (gfx1250) — matrix operand order repaired and checked by reading, not compiled, not run on hardware
//
#include <hip/hip_runtime.h>
#include <hip/hip_bf16.h>

// ---- sizes (compile-time, from reference) ----
#define BS   64
#define PP   512
#define DD   1024
#define DWM  1024
#define HH   16
#define WW   128
#define HD   64          // DWM / HH
#define MROWS (BS*PP)    // 32768
// SCALE = 1/sqrt(64) = 0.125f

typedef __bf16 bf16;
typedef __attribute__((ext_vector_type(16))) bf16 v16bf;
typedef __attribute__((ext_vector_type(8)))  bf16 v8bf;
typedef __attribute__((ext_vector_type(8)))  float v8f;

// ---------------------------------------------------------------------------
// f32 -> bf16 elementwise convert (row-major, used for A-side of GEMMs)
// ---------------------------------------------------------------------------
__global__ void wm_f32_to_bf16(const float* __restrict__ src,
                               bf16* __restrict__ dst, long long n) {
  long long i = (long long)blockIdx.x * blockDim.x + threadIdx.x;
  if (i < n) dst[i] = (bf16)src[i];
}

// ---------------------------------------------------------------------------
// Pack a row-major f32 weight W[K][N] into per-(ktile, ntile, lane) bf16
// B-matrix fragments for v_wmma_f32_16x16x32_bf16.
// Fragment element e (0..15) of lane l = W[k0 + (l>>4)*16 + e][n0 + (l&15)].
// Packed offset = ((kt*(N/16) + nb)*32 + lane)*16.
// ---------------------------------------------------------------------------
__global__ void wm_pack_weight(const float* __restrict__ W,
                               bf16* __restrict__ Wp, int K, int N) {
  int idx = blockIdx.x * blockDim.x + threadIdx.x;
  int total = (K >> 5) * (N >> 4) * 32;
  if (idx >= total) return;
  int lane = idx & 31;
  int grp  = idx >> 5;             // kt*(N/16) + nb
  int nb   = grp % (N >> 4);
  int kt   = grp / (N >> 4);
  int n    = nb * 16 + (lane & 15);
  int koff = kt * 32 + (lane >> 4) * 16;
  bf16* dst = Wp + (size_t)idx * 16;
#pragma unroll
  for (int e = 0; e < 16; ++e)
    dst[e] = (bf16)W[(size_t)(koff + e) * N + n];
}

// ---------------------------------------------------------------------------
// bf16 WMMA GEMM:  C[M][N] = A[M][K] (row-major bf16) x Bp (packed bf16) + bias
// Block = 128 threads = 4 waves; wave tile 64x64 (4x4 x 16x16), block 128x128.
// ---------------------------------------------------------------------------
__global__ __launch_bounds__(128)
void wm_gemm_bf16(const bf16* __restrict__ A, const bf16* __restrict__ Bp,
                  const float* __restrict__ bias, float* __restrict__ C,
                  int M, int N, int K) {
  const int lane = threadIdx.x & 31;
  const int wave = threadIdx.x >> 5;
  const int m0   = blockIdx.y * 128 + (wave >> 1) * 64;
  const int n0   = blockIdx.x * 128 + (wave & 1) * 64;
  const int mrow = lane & 15;     // A row within 16-row subtile
  const int kg   = lane >> 4;     // A k-group (0/1)
  const int nb16 = N >> 4;

  v8f acc[4][4];
#pragma unroll
  for (int mi = 0; mi < 4; ++mi)
#pragma unroll
    for (int ni = 0; ni < 4; ++ni) {
      v8f z = {0.f,0.f,0.f,0.f,0.f,0.f,0.f,0.f};
      acc[mi][ni] = z;
    }

  const int nK = K >> 5;
  for (int kt = 0; kt < nK; ++kt) {
    const int k0 = kt << 5;
    v16bf a[4], b[4];
#pragma unroll
    for (int mi = 0; mi < 4; ++mi) {
      const bf16* ap = A + (size_t)(m0 + mi * 16 + mrow) * K + k0 + kg * 16;
      v8bf lo = *(const v8bf*)ap;          // K = k0 + kg*16 + 0..7
      v8bf hi = *(const v8bf*)(ap + 8);    // K = k0 + kg*16 + 8..15
      a[mi] = __builtin_shufflevector(lo, hi, 0,1,2,3,4,5,6,7,
                                             8,9,10,11,12,13,14,15);
    }
#pragma unroll
    for (int ni = 0; ni < 4; ++ni)
      b[ni] = *(const v16bf*)(Bp + ((size_t)kt * nb16 + (n0 >> 4) + ni) * 512
                                 + lane * 16);
#pragma unroll
    for (int mi = 0; mi < 4; ++mi)
#pragma unroll
      for (int ni = 0; ni < 4; ++ni)
        acc[mi][ni] = __builtin_amdgcn_wmma_f32_16x16x32_bf16(
            false, a[mi], false, b[ni], (short)0, acc[mi][ni], false, false);
  }

  // C/D layout: VGPR r -> row r (lanes 0-15) / r+8 (lanes 16-31), col = lane&15
#pragma unroll
  for (int mi = 0; mi < 4; ++mi)
#pragma unroll
    for (int ni = 0; ni < 4; ++ni) {
      const int col = n0 + ni * 16 + (lane & 15);
      const float bv = bias[col];
#pragma unroll
      for (int r = 0; r < 8; ++r) {
        const int row = m0 + mi * 16 + r + 8 * kg;
        C[(size_t)row * N + col] = acc[mi][ni][r] + bv;
      }
    }
}

// ---------------------------------------------------------------------------
// Sequential ring-buffer attention scan. One block per (batch, head);
// 128 threads = one per window slot. K kept f32 transposed [d][slot] in LDS
// (padded, conflict-free score reads), V in bf16. Per-step q/k/v loads are
// software-pipelined (registers hold step t while step t+1's loads are in
// flight) and prefetched 8 steps ahead (global_prefetch_b8) so the serial
// 512-step chain is not gated on HBM latency. Writes attention output
// (pre-Wo) directly as bf16 in [b][t][h*HD+d] layout for the final GEMM.
// ---------------------------------------------------------------------------
__global__ __launch_bounds__(128)
void wm_scan(const float* __restrict__ q_all, const float* __restrict__ k_all,
             const float* __restrict__ v_all,
             const unsigned char* __restrict__ reset,
             bf16* __restrict__ attn_out) {
  const int b = blockIdx.x >> 4;   // / HH
  const int h = blockIdx.x & 15;   // % HH
  const int w = threadIdx.x;       // window slot 0..127

  __shared__ float KshT[HD][WW + 1];   // [d][slot], pad -> conflict-free reads
  __shared__ bf16  Vsh[WW][HD];        // [slot][d]
  __shared__ float qsh[HD];
  __shared__ float probs[WW];
  __shared__ float red[4];

  bool valid = false;
  int  ptr   = 0;
  const size_t baseQ = (size_t)b * PP * DWM + (size_t)h * HD;

  // pipeline prologue: registers hold step 0's q/k/v
  float qv = 0.f, kv = 0.f, vv = 0.f;
  if (w < HD) {
    qv = q_all[baseQ + w];
    kv = k_all[baseQ + w];
    vv = v_all[baseQ + w];
  }

  for (int t = 0; t < PP; ++t) {
    if (reset[(size_t)b * PP + t]) { valid = false; ptr = 0; }
    const size_t off = baseQ + (size_t)t * DWM;

    if (w < HD) {
      qsh[w]       = qv;
      KshT[w][ptr] = kv;
      Vsh[ptr][w]  = (bf16)vv;
    }
    if (w == ptr) valid = true;   // newly written slot is valid
    __syncthreads();

    // issue next step's loads now; they retire during this step's compute
    if (w < HD && t + 1 < PP) {
      const size_t offn = off + DWM + w;
      qv = q_all[offn];
      kv = k_all[offn];
      vv = v_all[offn];
      if (t + 8 < PP) {           // deep prefetch into L2/L0
        const size_t offp = off + (size_t)8 * DWM + w;
        __builtin_prefetch(q_all + offp, 0, 1);
        __builtin_prefetch(k_all + offp, 0, 1);
        __builtin_prefetch(v_all + offp, 0, 1);
      }
    }

    // scores
    float s = -3.0e38f;
    if (valid) {
      float a = 0.f;
#pragma unroll 8
      for (int d = 0; d < HD; ++d) a += qsh[d] * KshT[d][w];
      s = a * 0.125f;   // 1/sqrt(64)
    }

    // block max (wave32 shuffle + 4-entry LDS relay)
    float m = s;
#pragma unroll
    for (int o = 16; o > 0; o >>= 1) m = fmaxf(m, __shfl_xor(m, o));
    if ((w & 31) == 0) red[w >> 5] = m;
    __syncthreads();
    m = fmaxf(fmaxf(red[0], red[1]), fmaxf(red[2], red[3]));
    __syncthreads();

    float p = valid ? __expf(s - m) : 0.f;
    float sum = p;
#pragma unroll
    for (int o = 16; o > 0; o >>= 1) sum += __shfl_xor(sum, o);
    if ((w & 31) == 0) red[w >> 5] = sum;
    __syncthreads();
    sum = red[0] + red[1] + red[2] + red[3];   // >0: new slot always valid

    probs[w] = p / sum;
    __syncthreads();

    if (w < HD) {
      float o2 = 0.f;
#pragma unroll 8
      for (int ww = 0; ww < WW; ++ww) o2 += probs[ww] * (float)Vsh[ww][w];
      attn_out[off + w] = (bf16)o2;
    }
    __syncthreads();             // protect LDS before next step's writes
    ptr = (ptr + 1) & (WW - 1);
  }
}

// ---------------------------------------------------------------------------
extern "C" void kernel_launch(void* const* d_in, const int* in_sizes, int n_in,
                              void* d_out, int out_size, void* d_ws, size_t ws_size,
                              hipStream_t stream) {
  const float*         x     = (const float*)d_in[0];
  const unsigned char* reset = (const unsigned char*)d_in[1];  // np.bool_ (1B)
  const float* Wq = (const float*)d_in[2];  const float* bq = (const float*)d_in[3];
  const float* Wk = (const float*)d_in[4];  const float* bk = (const float*)d_in[5];
  const float* Wv = (const float*)d_in[6];  const float* bv = (const float*)d_in[7];
  const float* Wo = (const float*)d_in[8];  const float* bo = (const float*)d_in[9];
  float* out = (float*)d_out;

  // workspace carve-up (256B aligned chunks)
  char* ws = (char*)d_ws;
  size_t o = 0;
  bf16* x_bf  = (bf16*)(ws + o);  o += (size_t)MROWS * DD * 2;        // 64 MB
  bf16* Wq_p  = (bf16*)(ws + o);  o += (size_t)DD * DWM * 2;          // 2 MB
  bf16* Wk_p  = (bf16*)(ws + o);  o += (size_t)DD * DWM * 2;
  bf16* Wv_p  = (bf16*)(ws + o);  o += (size_t)DD * DWM * 2;
  bf16* Wo_p  = (bf16*)(ws + o);  o += (size_t)DWM * DD * 2;
  float* q_f  = (float*)(ws + o); o += (size_t)MROWS * DWM * 4;       // 128 MB
  float* k_f  = (float*)(ws + o); o += (size_t)MROWS * DWM * 4;
  float* v_f  = (float*)(ws + o); o += (size_t)MROWS * DWM * 4;
  bf16* attn  = (bf16*)(ws + o);  o += (size_t)MROWS * DWM * 2;       // 64 MB

  // 1) x -> bf16
  {
    long long n = (long long)MROWS * DD;
    int blk = 256;
    long long nb = (n + blk - 1) / blk;
    wm_f32_to_bf16<<<(unsigned)nb, blk, 0, stream>>>(x, x_bf, n);
  }

  // 2) pack weights into B-fragment order
  {
    int total = (DD >> 5) * (DWM >> 4) * 32;   // 65536
    int blk = 256, nb = (total + blk - 1) / blk;
    wm_pack_weight<<<nb, blk, 0, stream>>>(Wq, Wq_p, DD, DWM);
    wm_pack_weight<<<nb, blk, 0, stream>>>(Wk, Wk_p, DD, DWM);
    wm_pack_weight<<<nb, blk, 0, stream>>>(Wv, Wv_p, DD, DWM);
    wm_pack_weight<<<nb, blk, 0, stream>>>(Wo, Wo_p, DWM, DD);
  }

  // 3) Q/K/V projection GEMMs  [32768 x 1024] = [32768 x 1024] x [1024 x 1024]
  {
    dim3 grid(DWM / 128, MROWS / 128);   // (8, 256)
    wm_gemm_bf16<<<grid, 128, 0, stream>>>(x_bf, Wq_p, bq, q_f, MROWS, DWM, DD);
    wm_gemm_bf16<<<grid, 128, 0, stream>>>(x_bf, Wk_p, bk, k_f, MROWS, DWM, DD);
    wm_gemm_bf16<<<grid, 128, 0, stream>>>(x_bf, Wv_p, bv, v_f, MROWS, DWM, DD);
  }

  // 4) recurrent windowed attention over P steps
  wm_scan<<<BS * HH, WW, 0, stream>>>(q_f, k_f, v_f, reset, attn);

  // 5) output projection  out = attn @ Wo + bo
  {
    dim3 grid(DD / 128, MROWS / 128);
    wm_gemm_bf16<<<grid, 128, 0, stream>>>(attn, Wo_p, bo, out, MROWS, DD, DWM);
  }
}
